// TransformerBlock_62594853372292
// MI455X (gfx1250) — compile-verified, matrix operand order checked by reading, not run on hardware
//
#include <hip/hip_runtime.h>
#include <hip/hip_bf16.h>
#include <math.h>

// Problem constants (match reference).
#define BB 2
#define NN 512
#define NCC 128
#define ECC 64
#define HH 8
#define DD 16
#define ECP (ECC + 8)  // padded LDS row stride (halves): 144B, 16B aligned

typedef __attribute__((ext_vector_type(16))) _Float16 v16h;
typedef __attribute__((ext_vector_type(8)))  float    v8f;
typedef __attribute__((ext_vector_type(4))) unsigned int u32x4;
typedef __attribute__((ext_vector_type(8))) int i32x8;
typedef __attribute__((ext_vector_type(4))) int i32x4;

#if defined(__gfx1250__) && __has_builtin(__builtin_amdgcn_tensor_load_to_lds) && \
    __has_builtin(__builtin_amdgcn_s_wait_tensorcnt)
#define USE_TDM 1
#else
#define USE_TDM 0
#endif

// Compile-time probe signal (device pass only) so stderr tells us which path
// was compiled in.
#if defined(__AMDGCN__)
#if USE_TDM
#warning "CDNA5 probe: TDM tensor_load_to_lds path ENABLED in compute_G"
#else
#warning "CDNA5 probe: TDM builtin NOT available; manual LDS fill fallback"
#endif
#endif

__device__ __forceinline__ v8f wmma16(v16h a, v16h b, v8f c) {
  // D = A(16x32 f16) * B(32x16 f16) + C(16x16 f32)
  return __builtin_amdgcn_wmma_f32_16x16x32_f16(false, a, false, b, (short)0, c,
                                                false, false);
}

// K index inside a 16x32 A / 32x16 B fragment for element e of this lane.
__device__ __forceinline__ int kmap(int lane, int e) {
  return ((e < 8) ? e : e + 8) + ((lane & 16) ? 8 : 0);
}

// A fragment: A[m,k] = P[(mBase+lane%16)*ld + kBase + kmap]
__device__ __forceinline__ v16h load_a_f32(const float* __restrict__ P, int mBase,
                                           int kBase, int ld, int lane) {
  v16h a;
  int m = mBase + (lane & 15);
#pragma unroll
  for (int e = 0; e < 16; ++e) a[e] = (_Float16)P[(long)m * ld + kBase + kmap(lane, e)];
  return a;
}

// B fragment: B[k,n] = P[(kBase+kmap)*kStride + lane%16]
__device__ __forceinline__ v16h load_b_f32(const float* __restrict__ P, long kStride,
                                           int kBase, int lane) {
  v16h b;
  int n = lane & 15;
#pragma unroll
  for (int e = 0; e < 16; ++e)
    b[e] = (_Float16)P[(long)(kBase + kmap(lane, e)) * kStride + n];
  return b;
}

// A or B fragment from an LDS tile laid out [row][stride] f16 (row = M for A,
// row = N for B): per lane the 16 elements are two contiguous 8-half (16B)
// runs -> ds_load_b128 pairs.
__device__ __forceinline__ v16h load_frag_lds(const _Float16* base, int row, int stride,
                                              int kBase, int lane) {
  const _Float16* p = base + row * stride + kBase + ((lane & 16) ? 8 : 0);
  v16h v;
#pragma unroll
  for (int e = 0; e < 8; ++e) v[e] = p[e];
#pragma unroll
  for (int e = 8; e < 16; ++e) v[e] = p[e + 8];
  return v;
}

__device__ __forceinline__ float silu(float x) { return x / (1.0f + __expf(-x)); }

// ---------------------------------------------------------------------------
// K0: node_mask[b,i] = any_j edge_mask[b,i,j]
// ---------------------------------------------------------------------------
__global__ __launch_bounds__(256) void node_mask_k(const unsigned char* __restrict__ em,
                                                   float* __restrict__ nm) {
  int p = blockIdx.x * 256 + threadIdx.x;
  if (p >= BB * NN) return;
  const unsigned int* row = (const unsigned int*)(em + (long)p * NN);
  unsigned int any = 0;
#pragma unroll 4
  for (int j = 0; j < NN / 4; ++j) any |= row[j];
  nm[p] = any ? 1.0f : 0.0f;
}

// ---------------------------------------------------------------------------
// Generic node GEMM:  Out[b,m,i] = f(W[m,:]·Xin[b,:,i] + bias[m])
//   f: optional silu, optional *node_mask, optional +Res[b,m,i]
// Layouts: Xin (B,K,NN), Out/Res (B,M,NN). One wave = 16x16 output tile.
// ---------------------------------------------------------------------------
template <bool SILU, bool MASK, bool RES>
__global__ __launch_bounds__(128) void node_gemm(const float* __restrict__ W,
                                                 const float* __restrict__ bias,
                                                 const float* __restrict__ Xin,
                                                 const float* __restrict__ Res,
                                                 const float* __restrict__ nmask,
                                                 float* __restrict__ Out, int M, int K) {
  const int lane = threadIdx.x & 31;
  const int wv = threadIdx.x >> 5;
  const int wid = blockIdx.x * 4 + wv;
  const int mtiles = M >> 4;
  const int mt = wid % mtiles;
  const int pt = wid / mtiles;  // 0..63 position tiles over B*NN=1024
  const int pBase = pt << 4;
  const int n = lane & 15;
  const int p = pBase + n;
  const int pb = p >> 9;   // batch
  const int pi = p & 511;  // node index
  v8f acc = {};
  const float* xp = Xin + ((long)pb * K << 9) + pi;
  for (int ks = 0; ks < K; ks += 32) {
    v16h a = load_a_f32(W, mt * 16, ks, K, lane);
    v16h bf;
#pragma unroll
    for (int e = 0; e < 16; ++e) bf[e] = (_Float16)xp[(long)(ks + kmap(lane, e)) << 9];
    acc = wmma16(a, bf, acc);
  }
  const float msk = MASK ? nmask[p] : 1.0f;
  const int hi = (lane & 16) ? 8 : 0;
#pragma unroll
  for (int r = 0; r < 8; ++r) {
    int m = mt * 16 + r + hi;
    float v = acc[r] + bias[m];
    if (SILU) v = silu(v);
    if (MASK) v *= msk;
    long o = ((long)(pb * M + m) << 9) + pi;
    if (RES) v += Res[o];
    Out[o] = v;
  }
}

// ---------------------------------------------------------------------------
// K2: fused edge pipeline. One wave owns (b, i, 16 j's):
//   E_qk = qk_edge_w @ E  (WMMA, M=128,K=64)  -> E_q/E_k in registers
//   E_new = silu(E_k + silu(t)*E_q + t)*em, t = X_q[i]+X_k[j]  -> LDS + ws(f16)
//   E_out = E + (o_edge_w @ E_new + b)*em     (WMMA from LDS)  -> d_out
//   att   = att_w @ E_new + b, -inf where !em (WMMA from LDS)  -> ws logits
// ---------------------------------------------------------------------------
__global__ __launch_bounds__(256) void edge_phase1(
    const float* __restrict__ E, const unsigned char* __restrict__ emask,
    const float* __restrict__ Xqk, const float* __restrict__ qkw,
    const float* __restrict__ qkb, const float* __restrict__ ow,
    const float* __restrict__ ob, const float* __restrict__ aw,
    const float* __restrict__ ab, _Float16* __restrict__ Enew,
    float* __restrict__ attL, float* __restrict__ outE) {
  __shared__ _Float16 lds[8][16 * ECP];  // transposed [j][c] tile per wave
  const int lane = threadIdx.x & 31;
  const int wv = threadIdx.x >> 5;
  const int wid = blockIdx.x * 8 + wv;       // 32768 waves total
  const int jt = wid & 31;                   // 32 j-tiles
  const int i = (wid >> 5) & (NN - 1);
  const int b = wid >> 14;
  const int jBase = jt << 4;
  const int n = lane & 15;
  const int jcol = jBase + n;
  const int hi = (lane & 16) ? 8 : 0;
  const long rowC = (long)NN * NN;                         // channel stride in E
  const long base0 = (long)b * ECC * rowC + (long)i * NN;  // E[b,0,i,0]
  const float* Ebase = E + base0 + jBase;
  __builtin_prefetch(E + base0 + jcol, 0, 0);

  // --- E_qk (128 outputs = 8 WMMA M-tiles, K=64) ---
  float eq[4][8], ek[4][8];
#pragma unroll
  for (int mt = 0; mt < 8; ++mt) {
    v8f acc = {};
    acc = wmma16(load_a_f32(qkw, mt * 16, 0, ECC, lane), load_b_f32(Ebase, rowC, 0, lane), acc);
    acc = wmma16(load_a_f32(qkw, mt * 16, 32, ECC, lane), load_b_f32(Ebase, rowC, 32, lane), acc);
#pragma unroll
    for (int r = 0; r < 8; ++r) {
      int m = mt * 16 + r + hi;
      float v = acc[r] + qkb[m];
      if (mt < 4) eq[mt][r] = v;
      else        ek[mt - 4][r] = v;
    }
  }

  // --- elementwise gate -> E_new (LDS f16 + global f16) ---
  const float em01 = emask[((long)b * NN + i) * NN + jcol] ? 1.0f : 0.0f;
#pragma unroll
  for (int mt = 0; mt < 4; ++mt) {
#pragma unroll
    for (int r = 0; r < 8; ++r) {
      int c = mt * 16 + r + hi;
      float t = Xqk[(b * 2 * ECC + c) * NN + i] + Xqk[(b * 2 * ECC + ECC + c) * NN + jcol];
      float u = ek[mt][r] + silu(t) * eq[mt][r] + t;
      float v = silu(u) * em01;
      lds[wv][n * ECP + c] = (_Float16)v;
      Enew[base0 + (long)c * rowC + jcol] = (_Float16)v;
    }
  }
  __syncthreads();

  // --- E_out = E + (o_edge_w @ E_new + b)*em ---
#pragma unroll
  for (int mt = 0; mt < 4; ++mt) {
    v8f acc = {};
    acc = wmma16(load_a_f32(ow, mt * 16, 0, ECC, lane), load_frag_lds(lds[wv], n, ECP, 0, lane), acc);
    acc = wmma16(load_a_f32(ow, mt * 16, 32, ECC, lane), load_frag_lds(lds[wv], n, ECP, 32, lane), acc);
#pragma unroll
    for (int r = 0; r < 8; ++r) {
      int m = mt * 16 + r + hi;
      long o = base0 + (long)m * rowC + jcol;
      outE[o] = E[o] + (acc[r] + ob[m]) * em01;
    }
  }

  // --- attention logits (8 heads in one zero-padded 16-row tile) ---
  {
    v8f acc = {};
    const int mrow = lane & 15;
#pragma unroll
    for (int ks = 0; ks < 64; ks += 32) {
      v16h a;
#pragma unroll
      for (int e = 0; e < 16; ++e)
        a[e] = (mrow < 8) ? (_Float16)aw[mrow * ECC + ks + kmap(lane, e)] : (_Float16)0.0f;
      acc = wmma16(a, load_frag_lds(lds[wv], n, ECP, ks, lane), acc);
    }
    if (hi == 0) {
#pragma unroll
      for (int r = 0; r < 8; ++r) {
        float v = acc[r] + ab[r];
        attL[(((long)(b * HH + r)) * NN + i) * NN + jcol] =
            (em01 > 0.0f) ? v : -__builtin_inff();
      }
    }
  }
}

// ---------------------------------------------------------------------------
// K3: masked softmax over last axis; rows of length 512; write f16 probs.
// ---------------------------------------------------------------------------
__global__ __launch_bounds__(256) void softmax_rows(const float* __restrict__ logits,
                                                    _Float16* __restrict__ probs) {
  __shared__ float red[256];
  const long row = blockIdx.x;  // (b*H+h)*N + i, 8192 rows
  const float* p = logits + row * NN;
  const int t = threadIdx.x;
  float m = -__builtin_inff();
  for (int j = t; j < NN; j += 256) m = fmaxf(m, p[j]);
  red[t] = m;
  __syncthreads();
  for (int s = 128; s > 0; s >>= 1) {
    if (t < s) red[t] = fmaxf(red[t], red[t + s]);
    __syncthreads();
  }
  m = red[0];
  __syncthreads();
  float s = 0.0f;
  const bool ok = (m > -1e37f);
  for (int j = t; j < NN; j += 256) s += ok ? __expf(p[j] - m) : 0.0f;
  red[t] = s;
  __syncthreads();
  for (int k = 128; k > 0; k >>= 1) {
    if (t < k) red[t] += red[t + k];
    __syncthreads();
  }
  s = red[0];
  const float inv = (ok && s > 0.0f) ? 1.0f / s : 0.0f;  // nan_to_num path
  for (int j = t; j < NN; j += 256)
    probs[row * NN + j] = (_Float16)(ok ? __expf(p[j] - m) * inv : 0.0f);
}

// ---------------------------------------------------------------------------
// K4: G[b,h,e,i] = sum_j probs[b,h,i,j] * E_new[b,e,i,j]
// Per (b,i) this is a WMMA GEMM: G(64x8) = E_new(64x512) x att^T(512x8),
// heads padded to 16 columns (junk columns only affect discarded outputs,
// and the pad rows are zero-filled anyway).
// The 64x128 f16 E_new chunk (rows strided by N*N) is fetched with the
// Tensor Data Mover (2D D#: data_size=2B, tile 128x64, stride=N*N) into a
// double-buffered LDS tile, overlapped with the WMMA compute on the other
// buffer; manual-load fallback when the builtin is unavailable.
// ---------------------------------------------------------------------------
__global__ __launch_bounds__(128) void compute_G(const _Float16* __restrict__ probs,
                                                 const _Float16* __restrict__ Enew,
                                                 float* __restrict__ G) {
  __shared__ _Float16 attS[2][16 * 128];
  __shared__ _Float16 eS[2][ECC * 128];
  const int b = blockIdx.x >> 9;
  const int i = blockIdx.x & 511;
  const int t = threadIdx.x;
  const int lane = t & 31;
  const int wv = t >> 5;  // e-tile of this wave (rows wv*16..wv*16+15)

  auto loadAtt = [&](int buf, int jc) {
    for (int u = t; u < 16 * 128; u += 128) {
      int h = u >> 7, j = u & 127;
      attS[buf][u] = (h < HH)
                         ? probs[(((long)(b * HH + h) * NN + i) << 9) + jc + j]
                         : (_Float16)0.0f;
    }
  };
  auto loadE = [&](int buf, int jc) {
#if USE_TDM
    if (t < 32) {  // one TDM descriptor per block (TDM ignores EXEC)
      unsigned long long ga =
          (unsigned long long)(const void*)(Enew + (((long)(b * ECC) * NN + i) << 9) + jc);
      unsigned ldsOff = (unsigned)(unsigned long long)(const void*)&eS[buf][0];
      u32x4 g0;
      g0[0] = 1u;                                   // count=1, user descriptor
      g0[1] = ldsOff;                               // lds_addr (bytes)
      g0[2] = (unsigned)(ga & 0xffffffffu);         // global_addr[31:0]
      g0[3] = (unsigned)((ga >> 32) & 0x1ffffffu) | (2u << 30);  // addr hi | type=2
      i32x8 g1;
      unsigned long long s0 = (unsigned long long)NN * NN;  // dim0 stride (elems)
      g1[0] = (int)(1u << 16);                      // data_size = 2 bytes
      g1[1] = (int)((unsigned)NN << 16);            // tensor_dim0 = 512
      g1[2] = (int)((unsigned)ECC << 16);           // tensor_dim1 = 64
      g1[3] = (int)(128u << 16);                    // tile_dim0 = 128
      g1[4] = (int)ECC;                             // tile_dim1 = 64
      g1[5] = (int)(s0 & 0xffffffffu);              // tensor_dim0_stride lo
      g1[6] = (int)((s0 >> 32) & 0xffffu);          // tensor_dim0_stride hi
      g1[7] = 0;
      i32x4 z = {0, 0, 0, 0};
#if __clang_major__ >= 23
      i32x8 z8 = {};
      __builtin_amdgcn_tensor_load_to_lds(g0, g1, z, z, z8, 0);
#else
      __builtin_amdgcn_tensor_load_to_lds(g0, g1, z, z, 0);
#endif
    }
#else
    for (int u = t; u < ECC * 128; u += 128) {
      int e = u >> 7, j = u & 127;
      eS[buf][u] = Enew[(((long)(b * ECC + e) * NN + i) << 9) + jc + j];
    }
#endif
  };

  loadE(0, 0);
  loadAtt(0, 0);
  v8f acc = {};
  for (int c = 0; c < 4; ++c) {
    const int cur = c & 1;
#if USE_TDM
    __builtin_amdgcn_s_wait_tensorcnt(0);  // chunk c's DMA done (wave0); noop elsewhere
#endif
    __syncthreads();  // cur buffers ready; prev compute done before refilling
    if (c + 1 < 4) {
      loadE(cur ^ 1, (c + 1) * 128);
      loadAtt(cur ^ 1, (c + 1) * 128);
    }
#pragma unroll
    for (int ks = 0; ks < 128; ks += 32) {
      v16h a = load_frag_lds(eS[cur], wv * 16 + (lane & 15), 128, ks, lane);
      v16h bf = load_frag_lds(attS[cur], lane & 15, 128, ks, lane);
      acc = wmma16(a, bf, acc);
    }
  }
  const int h = lane & 15;
  const int hi = (lane & 16) ? 8 : 0;
  if (h < HH) {
#pragma unroll
    for (int r = 0; r < 8; ++r) {
      int e = wv * 16 + r + hi;
      G[((b * HH + h) * ECC + e) * NN + i] = acc[r];
    }
  }
}

// ---------------------------------------------------------------------------
// K5: X_new[b, h*16+d, i] = sum_j att*X_v  +  v_edge_w @ G + v_edge_b
// One wave = (b, h, 16 i's): D[d,i] tile; K=512 att pass + K=64 G pass.
// ---------------------------------------------------------------------------
__global__ __launch_bounds__(128) void attn_apply(const _Float16* __restrict__ probs,
                                                  const float* __restrict__ Xv,
                                                  const float* __restrict__ G,
                                                  const float* __restrict__ vw,
                                                  const float* __restrict__ vb,
                                                  float* __restrict__ Xnew) {
  const int lane = threadIdx.x & 31;
  const int wv = threadIdx.x >> 5;
  const int wid = blockIdx.x * 4 + wv;  // 512 waves
  const int it = wid & 31;
  const int h = (wid >> 5) & 7;
  const int b = wid >> 8;
  const int iBase = it << 4;
  const int n = lane & 15;
  v8f acc = {};
  // att (i x j) x X_v (j x d):  A[d,k]=Xv[b,h*16+d,k], B[k,i]=probs[b,h,iBase+i,k]
  const float* xvp = Xv + ((long)(b * NCC + h * DD) << 9);
  const _Float16* pp = probs + (((long)(b * HH + h) * NN + iBase + n) << 9);
  for (int ks = 0; ks < NN; ks += 32) {
    v16h a = load_a_f32(xvp, 0, ks, NN, lane);
    v16h bf;
#pragma unroll
    for (int e = 0; e < 16; ++e) bf[e] = pp[ks + kmap(lane, e)];
    acc = wmma16(a, bf, acc);
  }
  // + v_edge_w[h*16+d, e] * G[b,h,e,i]
  const float* gp = G + ((long)(b * HH + h) * ECC << 9) + iBase;
#pragma unroll
  for (int ks = 0; ks < ECC; ks += 32)
    acc = wmma16(load_a_f32(vw + (h * DD) * ECC, 0, ks, ECC, lane),
                 load_b_f32(gp, NN, ks, lane), acc);
  const int hi = (lane & 16) ? 8 : 0;
#pragma unroll
  for (int r = 0; r < 8; ++r) {
    int c = h * DD + r + hi;
    Xnew[((long)(b * NCC + c) << 9) + iBase + n] = acc[r] + vb[c];
  }
}

// ---------------------------------------------------------------------------
extern "C" void kernel_launch(void* const* d_in, const int* in_sizes, int n_in,
                              void* d_out, int out_size, void* d_ws, size_t ws_size,
                              hipStream_t stream) {
  const float* X = (const float*)d_in[0];
  const float* E = (const float*)d_in[1];
  const unsigned char* emask = (const unsigned char*)d_in[2];
  const float* qk_node_w = (const float*)d_in[3];
  const float* qk_node_b = (const float*)d_in[4];
  const float* qk_edge_w = (const float*)d_in[5];
  const float* qk_edge_b = (const float*)d_in[6];
  const float* v_node_w = (const float*)d_in[7];
  const float* v_node_b = (const float*)d_in[8];
  const float* v_edge_w = (const float*)d_in[9];
  const float* v_edge_b = (const float*)d_in[10];
  const float* o_node_w = (const float*)d_in[11];
  const float* o_node_b = (const float*)d_in[12];
  const float* o_edge_w = (const float*)d_in[13];
  const float* o_edge_b = (const float*)d_in[14];
  const float* att_w = (const float*)d_in[15];
  const float* att_b = (const float*)d_in[16];
  const float* mlp_w1 = (const float*)d_in[17];
  const float* mlp_b1 = (const float*)d_in[18];
  const float* mlp_w2 = (const float*)d_in[19];
  const float* mlp_b2 = (const float*)d_in[20];

  float* outX = (float*)d_out;               // (B,NC,N)
  float* outE = outX + (long)BB * NCC * NN;  // (B,EC,N,N)

  char* w = (char*)d_ws;
  auto alloc = [&](size_t bytes) -> char* {
    char* p = w;
    w += (bytes + 255) & ~(size_t)255;
    return p;
  };
  float* nm   = (float*)alloc((size_t)BB * NN * 4);
  float* Xqk  = (float*)alloc((size_t)BB * 2 * ECC * NN * 4);
  float* Xv   = (float*)alloc((size_t)BB * NCC * NN * 4);
  float* Xnew = (float*)alloc((size_t)BB * NCC * NN * 4);
  float* Hmlp = (float*)alloc((size_t)BB * 2 * NCC * NN * 4);
  float* G    = (float*)alloc((size_t)BB * HH * ECC * NN * 4);
  float* attL = (float*)alloc((size_t)BB * HH * NN * NN * 4);
  _Float16* probs = (_Float16*)alloc((size_t)BB * HH * NN * NN * 2);
  _Float16* Enew  = (_Float16*)alloc((size_t)BB * ECC * NN * NN * 2);

  // K0: node mask
  node_mask_k<<<(BB * NN + 255) / 256, 256, 0, stream>>>(emask, nm);
  // K1: X_qk = qk_node_w@X + b ; X_v = (v_node_w@X + b)*nm
  node_gemm<false, false, false><<<128, 128, 0, stream>>>(qk_node_w, qk_node_b, X,
                                                          nullptr, nullptr, Xqk, 128, 128);
  node_gemm<false, true, false><<<128, 128, 0, stream>>>(v_node_w, v_node_b, X,
                                                         nullptr, nm, Xv, 128, 128);
  // K2: fused edge pipeline (E_qk -> E_new -> E_out + att logits)
  edge_phase1<<<4096, 256, 0, stream>>>(E, emask, Xqk, qk_edge_w, qk_edge_b,
                                        o_edge_w, o_edge_b, att_w, att_b,
                                        Enew, attL, outE);
  // K3: softmax
  softmax_rows<<<BB * HH * NN, 256, 0, stream>>>(attL, probs);
  // K4: G reduction as WMMA GEMM; TDM double-buffered when available
  compute_G<<<BB * NN, 128, 0, stream>>>(probs, Enew, G);
  // K5: X_new = att@X_v + v_edge_w@G + v_edge_b
  attn_apply<<<128, 128, 0, stream>>>(probs, Xv, G, v_edge_w, v_edge_b, Xnew);
  // K6: Xo = X + (o_node_w@X_new + b)*nm
  node_gemm<false, true, true><<<128, 128, 0, stream>>>(o_node_w, o_node_b, Xnew,
                                                        X, nm, outX, 128, 128);
  // K7: h = silu(mlp_w1@Xo + b1)
  node_gemm<true, false, false><<<256, 128, 0, stream>>>(mlp_w1, mlp_b1, outX,
                                                         nullptr, nullptr, Hmlp, 256, 128);
  // K8: Xo = Xo + (mlp_w2@h + b2)*nm
  node_gemm<false, true, true><<<128, 128, 0, stream>>>(mlp_w2, mlp_b2, Hmlp,
                                                        outX, nm, outX, 128, 256);
}
